// NvidiaRelPartialLearnableMultiHeadAttn_32006096290239
// MI455X (gfx1250) — compile-verified, matrix operand order checked by reading, not run on hardware
//
#include <hip/hip_runtime.h>
#include <math.h>
#include <stdint.h>

// ---------------- shapes ----------------
#define QLEN   2048
#define BATCH  2
#define NH     16
#define DH     64
#define DM     1024
#define NQKV   3072
#define MROWS  (QLEN*BATCH)     // 4096, row = q*2+b
#define NEGINF (-__builtin_inff())

typedef __attribute__((ext_vector_type(16))) __bf16 v16bf;
typedef __attribute__((ext_vector_type(8)))  __bf16 v8bf;
typedef __attribute__((ext_vector_type(8)))  float  v8f;
typedef __attribute__((ext_vector_type(4)))  unsigned int u32x4;
typedef __attribute__((ext_vector_type(8)))  int i32x8;
typedef __attribute__((ext_vector_type(4)))  int i32x4;

// ---------------- helpers ----------------
__device__ inline v8f vzero8f() {
  v8f z;
#pragma unroll
  for (int i = 0; i < 8; ++i) z[i] = 0.0f;
  return z;
}

// Build a 16x32 bf16 A/B fragment half-pair: 8 bf16 from lo, 8 bf16 from hi.
__device__ inline v16bf make_frag(const __bf16* lo, const __bf16* hi) {
  v8bf a = *(const v8bf*)lo;
  v8bf b = *(const v8bf*)hi;
  v16bf r;
#pragma unroll
  for (int i = 0; i < 8; ++i) { r[i] = a[i]; r[i + 8] = b[i]; }
  return r;
}

__device__ inline v8f wmma_bf16(v16bf a, v16bf b, v8f c) {
  // (neg_a, A, neg_b, B, c_mod, C, reuse_a, reuse_b)
  return __builtin_amdgcn_wmma_f32_16x16x32_bf16(false, a, false, b, (short)0, c,
                                                 false, false);
}

__device__ inline float rowmax16(float x) {
#pragma unroll
  for (int m = 1; m < 16; m <<= 1) x = fmaxf(x, __shfl_xor(x, m, 32));
  return x;
}
__device__ inline float rowsum16(float x) {
#pragma unroll
  for (int m = 1; m < 16; m <<= 1) x += __shfl_xor(x, m, 32);
  return x;
}

// ---------------- Tensor Data Mover: 2D tile (bf16) global -> LDS ----------
// D# per CDNA5 ISA ch.8: group0 {count, lds_addr, global_addr, type=2},
// group1 {data_size=2B, tensor dims, tile dims, dim0 stride}, groups2/3 = 0.
__device__ inline void tdm_load_2d(unsigned lds_off, const void* gptr,
                                   unsigned tdim0, unsigned tdim1,
                                   unsigned tile0, unsigned tile1,
                                   unsigned long long stride0 /*elements*/) {
  unsigned long long ga = (unsigned long long)(uintptr_t)gptr;
  u32x4 g0;
  g0[0] = 1u;                                            // count=1, user desc
  g0[1] = lds_off;                                       // lds_addr (bytes)
  g0[2] = (unsigned)(ga & 0xffffffffull);                // global_addr[31:0]
  g0[3] = (unsigned)((ga >> 32) & 0x1ffffffull)          // global_addr[56:32]
        | (2u << 30);                                    // type=2 ("image")
  i32x8 g1;
  g1[0] = (int)(1u << 16);                               // data_size=1 (2 bytes)
  g1[1] = (int)((tdim0 & 0xffffu) << 16);                // tensor_dim0[15:0]
  g1[2] = (int)((tdim0 >> 16) | ((tdim1 & 0xffffu) << 16));
  g1[3] = (int)((tdim1 >> 16) | (tile0 << 16));          // tile_dim0
  g1[4] = (int)tile1;                                    // tile_dim1 (tile_dim2=0)
  g1[5] = (int)(stride0 & 0xffffffffull);                // tensor_dim0_stride lo
  g1[6] = (int)((stride0 >> 32) & 0xffffull);            // stride hi (dim1 stride=0)
  g1[7] = 0;
  i32x4 gz;
  gz[0] = 0; gz[1] = 0; gz[2] = 0; gz[3] = 0;
#if __clang_major__ >= 23
  i32x8 gz8;
  gz8[0] = 0; gz8[1] = 0; gz8[2] = 0; gz8[3] = 0;
  gz8[4] = 0; gz8[5] = 0; gz8[6] = 0; gz8[7] = 0;
  __builtin_amdgcn_tensor_load_to_lds(g0, g1, gz, gz, gz8, 0);
#else
  __builtin_amdgcn_tensor_load_to_lds(g0, g1, gz, gz, 0);
#endif
}

// ---------------- conversion / transpose ----------------
__global__ void k_cvt_bf16(const float* __restrict__ in, __bf16* __restrict__ out, int n) {
  int i = blockIdx.x * blockDim.x + threadIdx.x;
  if (i < n) out[i] = (__bf16)in[i];
}

// in[R][C] (f32) -> out[C][R] (bf16)
__global__ void k_transpose_bf16(const float* __restrict__ in, __bf16* __restrict__ out,
                                 int R, int C) {
  int i = blockIdx.x * blockDim.x + threadIdx.x;
  if (i < R * C) {
    int r = i / C, c = i % C;
    out[(size_t)c * R + r] = (__bf16)in[i];
  }
}

// ---------------- TN GEMM: C[M,N] = A[M,K] * Bt[N,K]^T (bf16 in, f32 out) ----
// Block: 256 thr = 8 waves; block tile 128(M) x 128(N); wave tile 16x128.
// B panel (shared by all 8 waves) staged to LDS by the Tensor Data Mover,
// double-buffered in 128x64 k-chunks; A fragments stream from global.
// All 8 B fragments are loaded into distinct registers before the WMMA burst
// so DS loads clause up and WMMAs issue back-to-back with partial waits.
__global__ __launch_bounds__(256) void k_gemm_tn(const __bf16* __restrict__ A,
                                                 const __bf16* __restrict__ Bt,
                                                 float* __restrict__ C,
                                                 int M, int N, int K) {
  __shared__ __align__(16) __bf16 bsm[2][128 * 64];   // 2 x 16 KB
  const int lane = threadIdx.x & 31;
  const int wave = threadIdx.x >> 5;
  const int row0 = blockIdx.y * 128 + wave * 16;
  const int col0 = blockIdx.x * 128;
  const int r  = lane & 15;
  const int hh = lane >> 4;
  const int kb = hh * 8;

  v8f acc[8];
#pragma unroll
  for (int f = 0; f < 8; ++f) acc[f] = vzero8f();

  const __bf16* arow   = A  + (size_t)(row0 + r) * K;
  const __bf16* bpanel = Bt + (size_t)col0 * K;        // 128 rows x K

  if (wave == 0)
    tdm_load_2d((unsigned)(uintptr_t)&bsm[0][0], bpanel,
                (unsigned)K, 128u, 64u, 128u, (unsigned long long)K);

  int cur = 0;
  for (int kt = 0; kt < K; kt += 64) {
    if (wave == 0) __builtin_amdgcn_s_wait_tensorcnt(0);   // chunk `cur` landed
    __syncthreads();
    if (wave == 0 && kt + 64 < K)
      tdm_load_2d((unsigned)(uintptr_t)&bsm[cur ^ 1][0], bpanel + (kt + 64),
                  (unsigned)K, 128u, 64u, 128u, (unsigned long long)K);

    const __bf16* bs = &bsm[cur][0];
    __builtin_prefetch(arow + kt + 128, 0, 1);             // global_prefetch_b8
#pragma unroll
    for (int s = 0; s < 2; ++s) {                          // two 32-wide k-steps
      const __bf16* ap = arow + kt + s * 32;
      v16bf af = make_frag(ap + kb, ap + 16 + kb);
      v16bf bfr[8];
#pragma unroll
      for (int f = 0; f < 8; ++f) {
        const __bf16* bp = bs + (f * 16 + r) * 64 + s * 32;
        bfr[f] = make_frag(bp + kb, bp + 16 + kb);
      }
#pragma unroll
      for (int f = 0; f < 8; ++f)
        acc[f] = wmma_bf16(af, bfr[f], acc[f]);
    }
    __syncthreads();                                       // reads done before overwrite
    cur ^= 1;
  }

#pragma unroll
  for (int f = 0; f < 8; ++f)
#pragma unroll
    for (int v = 0; v < 8; ++v) {
      int row = row0 + hh * 8 + v;
      int col = col0 + f * 16 + r;
      C[(size_t)row * N + col] = acc[f][v];
    }
}

// ---------------- repack heads (fuse biases) ----------------
// wheads f32 [4096][3072], row = q*2+b; cols [Q|K|V], inner = n*64+d
__global__ void k_repack_qkv(const float* __restrict__ wheads,
                             const float* __restrict__ rwb,   // [16][64]
                             const float* __restrict__ rrb,   // [16][64]
                             __bf16* __restrict__ Qrw, __bf16* __restrict__ Qrr,
                             __bf16* __restrict__ Kh,  __bf16* __restrict__ Vt) {
  int idx = blockIdx.x * blockDim.x + threadIdx.x;          // over 4096*1024
  if (idx >= MROWS * DM) return;
  int d   = idx & 63;
  int n   = (idx >> 6) & 15;
  int row = idx >> 10;
  int b   = row & 1;
  int q   = row >> 1;
  size_t base = (size_t)row * NQKV + n * 64 + d;
  float qv = wheads[base];
  float kv = wheads[base + DM];
  float vv = wheads[base + 2 * DM];
  size_t hq = ((size_t)(b * NH + n) * QLEN + q) * DH + d;
  Qrw[hq] = (__bf16)(qv + rwb[n * 64 + d]);
  Qrr[hq] = (__bf16)(qv + rrb[n * 64 + d]);
  Kh[hq]  = (__bf16)kv;
  Vt[((size_t)(b * NH + n) * DH + d) * QLEN + q] = (__bf16)vv;
}

// rkf f32 [2048][1024] -> RKh bf16 [n][j][d]
__global__ void k_repack_rk(const float* __restrict__ rkf, __bf16* __restrict__ RKh) {
  int idx = blockIdx.x * blockDim.x + threadIdx.x;          // over 16*2048*64
  if (idx >= NH * QLEN * DH) return;
  int d = idx & 63;
  int j = (idx >> 6) & (QLEN - 1);
  int n = idx >> 17;
  RKh[idx] = (__bf16)(rkf[(size_t)j * DM + n * 64 + d]);
}

// ---------------- flash attention ----------------
// One wave per (b, n, 16-query tile). Online softmax, causal mask,
// AC from Qrw*K^T, BD from Qrr[m]*r_head_k[m]^T with the rel-shift head remap:
// batch 0 head n reads BD of head n+1 (head 15 gets no BD term); batch 1 is identity.
__global__ __launch_bounds__(256) void k_attn(const __bf16* __restrict__ Qrw,
                                              const __bf16* __restrict__ Qrr,
                                              const __bf16* __restrict__ Kh,
                                              const __bf16* __restrict__ Vt,
                                              const __bf16* __restrict__ RKh,
                                              __bf16* __restrict__ AV) {
  __shared__ __align__(16) __bf16 plds[8 * 512];   // per-wave 16x32 P tile
  const int lane = threadIdx.x & 31;
  const int wave = threadIdx.x >> 5;
  const int tile = blockIdx.x * 8 + wave;          // 0..4095
  const int b   = tile >> 11;                      // 2048 tiles per batch
  const int rem = tile & 2047;
  const int n   = rem >> 7;                        // 128 tiles per head
  const int i0  = (rem & 127) << 4;
  const bool has_bd = !(b == 0 && n == 15);
  int m = (b == 0) ? n + 1 : n;
  if (!has_bd) m = n;

  const __bf16* Qw = Qrw + (size_t)(b * NH + n) * QLEN * DH;
  const __bf16* Qr = Qrr + (size_t)(b * NH + m) * QLEN * DH;
  const __bf16* Kb = Kh  + (size_t)(b * NH + n) * QLEN * DH;
  const __bf16* Vb = Vt  + (size_t)(b * NH + n) * DH * QLEN;
  const __bf16* Rb = RKh + (size_t)m * QLEN * DH;

  const int r  = lane & 15;
  const int hh = lane >> 4;
  const int kb = hh * 8;

  // Query fragments (two 32-wide k-steps across d_head=64), loaded once.
  v16bf aw[2], ar[2];
#pragma unroll
  for (int s = 0; s < 2; ++s) {
    const __bf16* qp = Qw + (size_t)(i0 + r) * DH + s * 32;
    aw[s] = make_frag(qp + kb, qp + 16 + kb);
    qp = Qr + (size_t)(i0 + r) * DH + s * 32;
    ar[s] = make_frag(qp + kb, qp + 16 + kb);
  }

  v8f o[4];
#pragma unroll
  for (int f = 0; f < 4; ++f) o[f] = vzero8f();
  float mrow[8], lrow[8];
#pragma unroll
  for (int v = 0; v < 8; ++v) { mrow[v] = NEGINF; lrow[v] = 0.0f; }

  __bf16* pl = plds + wave * 512;
  const int jmax = i0 + 15;

  for (int j0 = 0; j0 <= jmax; j0 += 32) {
    v8f s0 = vzero8f(), s1 = vzero8f();
    // ---- score tile 0 (cols j0..j0+15): hoist loads, then WMMA burst ----
    {
      v16bf bk[2];
#pragma unroll
      for (int s = 0; s < 2; ++s) {
        const __bf16* kp = Kb + (size_t)(j0 + r) * DH + s * 32;
        bk[s] = make_frag(kp + kb, kp + 16 + kb);
      }
      s0 = wmma_bf16(aw[0], bk[0], s0);
      s0 = wmma_bf16(aw[1], bk[1], s0);
      if (has_bd) {
        v16bf br[2];
#pragma unroll
        for (int s = 0; s < 2; ++s) {
          const __bf16* rp = Rb + (size_t)(j0 + r) * DH + s * 32;
          br[s] = make_frag(rp + kb, rp + 16 + kb);
        }
        s0 = wmma_bf16(ar[0], br[0], s0);
        s0 = wmma_bf16(ar[1], br[1], s0);
      }
    }
    // ---- score tile 1 (cols j0+16..j0+31) ----
    const bool t1 = (j0 + 16) <= jmax;
    if (t1) {
      v16bf bk[2];
#pragma unroll
      for (int s = 0; s < 2; ++s) {
        const __bf16* kp = Kb + (size_t)(j0 + 16 + r) * DH + s * 32;
        bk[s] = make_frag(kp + kb, kp + 16 + kb);
      }
      s1 = wmma_bf16(aw[0], bk[0], s1);
      s1 = wmma_bf16(aw[1], bk[1], s1);
      if (has_bd) {
        v16bf br[2];
#pragma unroll
        for (int s = 0; s < 2; ++s) {
          const __bf16* rp = Rb + (size_t)(j0 + 16 + r) * DH + s * 32;
          br[s] = make_frag(rp + kb, rp + 16 + kb);
        }
        s1 = wmma_bf16(ar[0], br[0], s1);
        s1 = wmma_bf16(ar[1], br[1], s1);
      }
    }
    // ---- scale, causal mask, online softmax ----
#pragma unroll
    for (int v = 0; v < 8; ++v) {
      const int irow = i0 + hh * 8 + v;
      float x0 = s0[v] * 0.125f;                 // 1/sqrt(64)
      if (j0 + r > irow) x0 = NEGINF;
      float x1 = t1 ? s1[v] * 0.125f : NEGINF;
      if (j0 + 16 + r > irow) x1 = NEGINF;
      float nm = fmaxf(mrow[v], rowmax16(fmaxf(x0, x1)));
      float alpha = (mrow[v] == NEGINF) ? 0.0f : __expf(mrow[v] - nm);
      float p0 = (x0 == NEGINF) ? 0.0f : __expf(x0 - nm);
      float p1 = (x1 == NEGINF) ? 0.0f : __expf(x1 - nm);
      lrow[v] = lrow[v] * alpha + rowsum16(p0 + p1);
      mrow[v] = nm;
#pragma unroll
      for (int f = 0; f < 4; ++f) o[f][v] *= alpha;
      pl[(hh * 8 + v) * 32 + r]      = (__bf16)p0;
      pl[(hh * 8 + v) * 32 + 16 + r] = (__bf16)p1;
    }
    asm volatile("s_wait_dscnt 0x0" ::: "memory");
    // ---- P (16x32) -> A-fragment, accumulate P*V (loads hoisted) ----
    const __bf16* pr = pl + r * 32;
    v16bf pa = make_frag(pr + kb, pr + 16 + kb);
    v16bf bv[4];
#pragma unroll
    for (int f = 0; f < 4; ++f) {
      const __bf16* vp = Vb + (size_t)(f * 16 + r) * QLEN + j0;
      bv[f] = make_frag(vp + kb, vp + 16 + kb);
    }
#pragma unroll
    for (int f = 0; f < 4; ++f)
      o[f] = wmma_bf16(pa, bv[f], o[f]);
  }

  // epilogue: normalize by l, write attn_vec bf16 [row=q*2+b][n*64+d]
#pragma unroll
  for (int v = 0; v < 8; ++v) {
    const int irow = i0 + hh * 8 + v;
    const float inv = 1.0f / lrow[v];
#pragma unroll
    for (int f = 0; f < 4; ++f)
      AV[(size_t)(irow * 2 + b) * DM + n * 64 + f * 16 + r] = (__bf16)(o[f][v] * inv);
  }
}

// ---------------- residual + LayerNorm ----------------
__global__ __launch_bounds__(256) void k_ln(const float* __restrict__ content,
                                            const float* __restrict__ proj,
                                            const float* __restrict__ g,
                                            const float* __restrict__ be,
                                            float* __restrict__ out) {
  const int row = blockIdx.x;
  const int t = threadIdx.x;
  __shared__ float sa[256], sb[256];
  float a = 0.0f, b2 = 0.0f;
  for (int c = t; c < DM; c += 256) {
    float v = content[(size_t)row * DM + c] + proj[(size_t)row * DM + c];
    a += v; b2 += v * v;
  }
  sa[t] = a; sb[t] = b2;
  __syncthreads();
  for (int st = 128; st > 0; st >>= 1) {
    if (t < st) { sa[t] += sa[t + st]; sb[t] += sb[t + st]; }
    __syncthreads();
  }
  const float mu  = sa[0] * (1.0f / DM);
  const float var = sb[0] * (1.0f / DM) - mu * mu;
  const float inv = rsqrtf(var + 1e-5f);
  for (int c = t; c < DM; c += 256) {
    float v = content[(size_t)row * DM + c] + proj[(size_t)row * DM + c];
    out[(size_t)row * DM + c] = (v - mu) * inv * g[c] + be[c];
  }
}

// ---------------- launcher ----------------
extern "C" void kernel_launch(void* const* d_in, const int* in_sizes, int n_in,
                              void* d_out, int out_size, void* d_ws, size_t ws_size,
                              hipStream_t stream) {
  const float* content = (const float*)d_in[0];   // [2048,2,1024]
  const float* rel_pos = (const float*)d_in[1];   // [2048,1024]
  const float* r_w_b   = (const float*)d_in[2];   // [16,64]
  const float* r_r_b   = (const float*)d_in[3];   // [16,64]
  // d_in[4] = attn_mask (causal; computed analytically)
  const float* W_qkv   = (const float*)d_in[5];   // [1024,3072]
  const float* W_o     = (const float*)d_in[6];   // [1024,1024]
  const float* W_r     = (const float*)d_in[7];   // [1024,1024]
  const float* ln_g    = (const float*)d_in[8];
  const float* ln_b    = (const float*)d_in[9];
  float* out = (float*)d_out;

  char* w = (char*)d_ws;
  size_t off = 0;
  auto take = [&](size_t bytes) -> char* {
    char* p = w + off;
    off = (off + bytes + 255) & ~(size_t)255;
    return p;
  };
  float*  wheads = (float*)take((size_t)MROWS * NQKV * 4);          // 50.3 MB
  float*  rkf    = (float*)take((size_t)QLEN * DM * 4);             //  8.4 MB
  __bf16* Xbf    = (__bf16*)take((size_t)MROWS * DM * 2);
  __bf16* WqkvT  = (__bf16*)take((size_t)NQKV * DM * 2);
  __bf16* Rbf    = (__bf16*)take((size_t)QLEN * DM * 2);
  __bf16* WrT    = (__bf16*)take((size_t)DM * DM * 2);
  __bf16* WoT    = (__bf16*)take((size_t)DM * DM * 2);
  __bf16* Qrw    = (__bf16*)take((size_t)BATCH * NH * QLEN * DH * 2);
  __bf16* Qrr    = (__bf16*)take((size_t)BATCH * NH * QLEN * DH * 2);
  __bf16* Khp    = (__bf16*)take((size_t)BATCH * NH * QLEN * DH * 2);
  __bf16* Vtp    = (__bf16*)take((size_t)BATCH * NH * QLEN * DH * 2);
  __bf16* RKh    = (__bf16*)take((size_t)NH * QLEN * DH * 2);
  __bf16* AV     = (__bf16*)take((size_t)MROWS * DM * 2);
  float*  proj   = wheads;   // alias: wheads dead after repack

  const int T = 256;
  // 1) converts / transposes
  k_cvt_bf16<<<(MROWS * DM + T - 1) / T, T, 0, stream>>>(content, Xbf, MROWS * DM);
  k_cvt_bf16<<<(QLEN * DM + T - 1) / T, T, 0, stream>>>(rel_pos, Rbf, QLEN * DM);
  k_transpose_bf16<<<(DM * NQKV + T - 1) / T, T, 0, stream>>>(W_qkv, WqkvT, DM, NQKV);
  k_transpose_bf16<<<(DM * DM + T - 1) / T, T, 0, stream>>>(W_r, WrT, DM, DM);
  k_transpose_bf16<<<(DM * DM + T - 1) / T, T, 0, stream>>>(W_o, WoT, DM, DM);
  // 2) projections (WMMA GEMMs, TDM-staged 128-wide B panels)
  k_gemm_tn<<<dim3(NQKV / 128, MROWS / 128), T, 0, stream>>>(Xbf, WqkvT, wheads,
                                                             MROWS, NQKV, DM);
  k_gemm_tn<<<dim3(DM / 128, QLEN / 128), T, 0, stream>>>(Rbf, WrT, rkf,
                                                          QLEN, DM, DM);
  // 3) repack heads + biases
  k_repack_qkv<<<(MROWS * DM + T - 1) / T, T, 0, stream>>>(wheads, r_w_b, r_r_b,
                                                           Qrw, Qrr, Khp, Vtp);
  k_repack_rk<<<(NH * QLEN * DH + T - 1) / T, T, 0, stream>>>(rkf, RKh);
  // 4) flash attention (512 blocks x 8 waves; one wave per 16-query tile)
  k_attn<<<(BATCH * NH * (QLEN / 16)) / 8, T, 0, stream>>>(Qrw, Qrr, Khp, Vtp, RKh, AV);
  // 5) output projection (WMMA GEMM) + 6) residual+LN
  k_gemm_tn<<<dim3(DM / 128, MROWS / 128), T, 0, stream>>>(AV, WoT, proj,
                                                           MROWS, DM, DM);
  k_ln<<<MROWS, T, 0, stream>>>(content, proj, ln_g, ln_b, out);
  (void)in_sizes; (void)n_in; (void)out_size; (void)ws_size;
}
